// MultiheadAttention_36627481100428
// MI455X (gfx1250) — matrix operand order repaired and checked by reading, not compiled, not run on hardware
//
#include <hip/hip_runtime.h>
#include <cstdint>
#include <cstddef>

// ---------------- problem constants ----------------
#define Lq   256   // query length
#define Sk   512   // key length
#define NB   2     // batch
#define DIMk 256   // model dim
#define NH   8     // heads
#define HDm  32    // head dim
#define HID  128   // MLP hidden
#define NCH  64    // n-chunk staged in LDS

typedef __bf16 bf16;
typedef __attribute__((ext_vector_type(16))) __bf16 v16bf;
typedef __attribute__((ext_vector_type(8)))  float  v8f;

__device__ __forceinline__ bf16 f2bf(float f) {
    uint32_t u = __builtin_bit_cast(uint32_t, f);
    u += 0x7fffu + ((u >> 16) & 1u);           // round-to-nearest-even
    unsigned short s = (unsigned short)(u >> 16);
    return __builtin_bit_cast(bf16, s);
}
__device__ __forceinline__ float bf2f(bf16 b) {
    unsigned short s = __builtin_bit_cast(unsigned short, b);
    uint32_t u = ((uint32_t)s) << 16;
    return __builtin_bit_cast(float, u);
}
// A-matrix (16x32 bf16) K index for element e, lane-group g (contiguous: k = 16g + e, matches B)
__device__ __forceinline__ int a_koff(int e, int g) {
    return e + (g ? 16 : 0);
}

// ---------------- elementwise convert ----------------
__global__ void k_f32_to_bf16(const float* __restrict__ src, bf16* __restrict__ dst, int n) {
    for (int i = blockIdx.x * blockDim.x + threadIdx.x; i < n; i += gridDim.x * blockDim.x)
        dst[i] = f2bf(src[i]);
}

// ---------------- generic WMMA GEMM: C[M,N] = A[M,K]bf16 @ Bt[N,K]bf16^T (+bias) ----------------
// one wave per 16x16 output tile
__global__ void __launch_bounds__(32) k_gemm_bf16(const bf16* __restrict__ A,
                                                  const bf16* __restrict__ Bt,
                                                  const float* __restrict__ bias,
                                                  float* __restrict__ C,
                                                  int M, int N, int K) {
    int lane = threadIdx.x & 31;
    int g = lane >> 4, col = lane & 15, row = lane & 15;
    int n0 = blockIdx.x * 16, m0 = blockIdx.y * 16;
    v8f c = {};
    for (int kk = 0; kk < K; kk += 32) {
        v16bf a, b;
#pragma unroll
        for (int e = 0; e < 16; ++e) a[e] = A[(size_t)(m0 + row) * K + kk + a_koff(e, g)];
#pragma unroll
        for (int e = 0; e < 16; ++e) b[e] = Bt[(size_t)(n0 + col) * K + kk + 16 * g + e];
        c = __builtin_amdgcn_wmma_f32_16x16x32_bf16(false, a, false, b, (short)0, c, false, false);
    }
#pragma unroll
    for (int r = 0; r < 8; ++r) {
        int mm = m0 + r + 8 * g;
        float v = c[r];
        if (bias) v += bias[n0 + col];
        C[(size_t)mm * N + n0 + col] = v;
    }
}

// ---------------- (t,b,dim) flat -> per-head layouts ----------------
__global__ void k_reshape_heads(const float* __restrict__ flat, float* __restrict__ fdst,
                                bf16* __restrict__ bdst, int T, float scale) {
    int total = NB * NH * T * HDm;
    for (int idx = blockIdx.x * blockDim.x + threadIdx.x; idx < total;
         idx += gridDim.x * blockDim.x) {
        int c = idx % HDm;
        int t = (idx / HDm) % T;
        int h = (idx / (HDm * T)) % NH;
        int b = idx / (HDm * T * NH);
        float v = flat[(size_t)(t * NB + b) * DIMk + h * HDm + c] * scale;
        if (fdst) fdst[idx] = v;
        if (bdst) bdst[idx] = f2bf(v);
    }
}

// ---------------- QW[b][m][16][HID] = qs(h,:) @ rq_w2(slice)  (rows 8..15 zero) ----------------
__global__ void k_qw(const float* __restrict__ qs, const float* __restrict__ w2,
                     bf16* __restrict__ QW) {
    int total = NB * Lq * 16 * HID;
    for (int idx = blockIdx.x * blockDim.x + threadIdx.x; idx < total;
         idx += gridDim.x * blockDim.x) {
        int j = idx % HID;
        int r = (idx / HID) % 16;
        int bm = idx / (HID * 16);
        int b = bm / Lq, m = bm % Lq;
        float acc = 0.f;
        if (r < NH) {
            const float* q = qs + ((size_t)(b * NH + r) * Lq + m) * HDm;
            for (int c = 0; c < HDm; ++c) acc = fmaf(q[c], w2[(r * HDm + c) * HID + j], acc);
        }
        QW[idx] = f2bf(acc);
    }
}

// ---------------- KW[b][n][h][HID] f32 = (kh*scale) @ rk_w2(slice) ----------------
__global__ void k_kw(const float* __restrict__ khf, const float* __restrict__ w2,
                     float* __restrict__ KW, float scale) {
    int total = NB * Sk * NH * HID;
    for (int idx = blockIdx.x * blockDim.x + threadIdx.x; idx < total;
         idx += gridDim.x * blockDim.x) {
        int j = idx % HID;
        int h = (idx / HID) % NH;
        int bn = idx / (HID * NH);
        int b = bn / Sk, n = bn % Sk;
        const float* kr = khf + ((size_t)(b * NH + h) * Sk + n) * HDm;
        float acc = 0.f;
        for (int c = 0; c < HDm; ++c) acc = fmaf(kr[c] * scale, w2[(h * HDm + c) * HID + j], acc);
        KW[idx] = acc;
    }
}

// ---------------- scalar bias terms qb[(b*L+m)*H+h], kb[(b*H+h)*S+n] ----------------
__global__ void k_qb(const float* __restrict__ qs, const float* __restrict__ b2,
                     float* __restrict__ qb) {
    int total = NB * Lq * NH;
    for (int idx = blockIdx.x * blockDim.x + threadIdx.x; idx < total;
         idx += gridDim.x * blockDim.x) {
        int h = idx % NH;
        int m = (idx / NH) % Lq;
        int b = idx / (NH * Lq);
        const float* q = qs + ((size_t)(b * NH + h) * Lq + m) * HDm;
        float acc = 0.f;
        for (int c = 0; c < HDm; ++c) acc = fmaf(q[c], b2[h * HDm + c], acc);
        qb[idx] = acc;
    }
}
__global__ void k_kb(const float* __restrict__ khf, const float* __restrict__ b2,
                     float* __restrict__ kb, float scale) {
    int total = NB * NH * Sk;
    for (int idx = blockIdx.x * blockDim.x + threadIdx.x; idx < total;
         idx += gridDim.x * blockDim.x) {
        int n = idx % Sk;
        int h = (idx / Sk) % NH;
        int b = idx / (Sk * NH);
        const float* kr = khf + ((size_t)(b * NH + h) * Sk + n) * HDm;
        float acc = 0.f;
        for (int c = 0; c < HDm; ++c) acc = fmaf(kr[c] * scale, b2[h * HDm + c], acc);
        kb[idx] = acc;
    }
}

// ---------------- logits + softmax, one block per (b,m). Hiddens computed in LDS chunks. ----------------
__global__ void __launch_bounds__(256) k_logits_softmax(
    const float* __restrict__ xyz,
    const float* __restrict__ rq_w1, const float* __restrict__ rq_b1,
    const float* __restrict__ rk_w1, const float* __restrict__ rk_b1,
    const bf16* __restrict__ QW, const float* __restrict__ qb,
    const float* __restrict__ KW, const float* __restrict__ kb,
    const float* __restrict__ qs, const float* __restrict__ khf,
    bf16* __restrict__ abf_hm, bf16* __restrict__ abf_mh) {
    __shared__ bf16  sHq[NCH][HID];    // 16 KB
    __shared__ bf16  sHk[NCH][HID];    // 16 KB
    __shared__ float sLog[NH][Sk];     // 16 KB
    __shared__ float sW[2][HID][4];    // layer-1 weights+bias for q/k MLPs
    __shared__ float sXYZ[NCH][3];

    int tid = threadIdx.x, lane = tid & 31, wave = tid >> 5;
    int bm = blockIdx.x;
    int b = bm / Lq, m = bm % Lq;

    for (int idx = tid; idx < 2 * HID; idx += 256) {
        int which = idx / HID, j = idx % HID;
        const float* w1 = which ? rk_w1 : rq_w1;
        const float* b1 = which ? rk_b1 : rq_b1;
        sW[which][j][0] = w1[j * 3 + 0];
        sW[which][j][1] = w1[j * 3 + 1];
        sW[which][j][2] = w1[j * 3 + 2];
        sW[which][j][3] = b1[j];
    }
    __syncthreads();

    const bf16* QWp = QW + (size_t)(b * Lq + m) * 16 * HID;

    for (int t = 0; t < Sk / NCH; ++t) {
        int n0 = t * NCH;
        for (int nn = tid; nn < NCH; nn += 256) {
            const float* p = xyz + ((size_t)(m * Sk + n0 + nn) * NB + b) * 3;
            sXYZ[nn][0] = p[0]; sXYZ[nn][1] = p[1]; sXYZ[nn][2] = p[2];
        }
        __syncthreads();
        // MLP layer-1 + ReLU into LDS (q and k hiddens)
        for (int idx = tid; idx < 2 * NCH * HID; idx += 256) {
            int which = idx / (NCH * HID);
            int r = idx % (NCH * HID);
            int nn = r / HID, j = r % HID;
            const float* w = sW[which][j];
            float hsum = fmaf(sXYZ[nn][0], w[0],
                         fmaf(sXYZ[nn][1], w[1],
                         fmaf(sXYZ[nn][2], w[2], w[3])));
            bf16 hv = f2bf(fmaxf(hsum, 0.f));
            if (which) sHk[nn][j] = hv; else sHq[nn][j] = hv;
        }
        __syncthreads();
        // q-position bias: (16 heads-padded x 128) @ (128 x 64) via WMMA, waves 0..3
        if (wave < 4) {
            int g = lane >> 4, col = lane & 15, row = lane & 15;
            v8f c = {};
#pragma unroll
            for (int kk = 0; kk < HID; kk += 32) {
                v16bf a, bfr;
#pragma unroll
                for (int e = 0; e < 16; ++e) a[e] = QWp[row * HID + kk + a_koff(e, g)];
#pragma unroll
                for (int e = 0; e < 16; ++e) bfr[e] = sHq[16 * wave + col][kk + 16 * g + e];
                c = __builtin_amdgcn_wmma_f32_16x16x32_bf16(false, a, false, bfr, (short)0, c,
                                                            false, false);
            }
#pragma unroll
            for (int r = 0; r < 8; ++r) {
                int h = r + 8 * g;
                if (h < NH) sLog[h][n0 + 16 * wave + col] = c[r];
            }
        }
        __syncthreads();
        // content dot + k-position bias + scalar biases (all waves, VALU)
        for (int idx = tid; idx < NH * NCH; idx += 256) {
            int h = idx / NCH, nn = idx % NCH, n = n0 + nn;
            const float* qrow = qs + ((size_t)(b * NH + h) * Lq + m) * HDm;
            const float* krow = khf + ((size_t)(b * NH + h) * Sk + n) * HDm;
            float acc = 0.f;
#pragma unroll
            for (int cc = 0; cc < HDm; ++cc) acc = fmaf(qrow[cc], krow[cc], acc);
            const float* kwrow = KW + ((size_t)(b * Sk + n) * NH + h) * HID;
            float bk = 0.f;
            for (int j = 0; j < HID; ++j) bk = fmaf(kwrow[j], bf2f(sHk[nn][j]), bk);
            acc += bk + qb[(size_t)(b * Lq + m) * NH + h] + kb[(size_t)(b * NH + h) * Sk + n];
            sLog[h][n] += acc;
        }
        __syncthreads();
    }
    // softmax: wave w handles head w
    {
        int h = wave;
        float mx = -3.0e38f;
        for (int n = lane; n < Sk; n += 32) mx = fmaxf(mx, sLog[h][n]);
#pragma unroll
        for (int o = 16; o > 0; o >>= 1) mx = fmaxf(mx, __shfl_xor(mx, o, 32));
        float sum = 0.f;
        for (int n = lane; n < Sk; n += 32) {
            float e = __expf(sLog[h][n] - mx);
            sLog[h][n] = e;
            sum += e;
        }
#pragma unroll
        for (int o = 16; o > 0; o >>= 1) sum += __shfl_xor(sum, o, 32);
        float inv = 1.f / sum;
        for (int n = lane; n < Sk; n += 32) {
            bf16 ab = f2bf(sLog[h][n] * inv);
            abf_hm[((size_t)(b * NH + h) * Lq + m) * Sk + n] = ab;
            abf_mh[((size_t)(b * Lq + m) * 16 + h) * Sk + n] = ab;
        }
    }
    // zero the pad rows (8..15) of the head-major A matrix
    for (int idx = tid; idx < 8 * Sk; idx += 256) {
        int r = 8 + idx / Sk, n = idx % Sk;
        abf_mh[((size_t)(b * Lq + m) * 16 + r) * Sk + n] = f2bf(0.f);
    }
}

// ---------------- x1 = a @ vh per (b,h): (16m x 512) @ (512 x 32) WMMA ----------------
__global__ void __launch_bounds__(64) k_x1(const bf16* __restrict__ abf_hm,
                                           const bf16* __restrict__ vhb,
                                           float* __restrict__ x1) {
    int lane = threadIdx.x & 31, wave = threadIdx.x >> 5;
    int g = lane >> 4, col = lane & 15, row = lane & 15;
    int idx = blockIdx.x;
    int mt = idx % (Lq / 16);
    int bh = idx / (Lq / 16);
    int b = bh / NH, h = bh % NH;
    const bf16* Ap = abf_hm + ((size_t)(b * NH + h) * Lq + 16 * mt) * Sk;
    const bf16* Bp = vhb + (size_t)(b * NH + h) * Sk * HDm;
    v8f c = {};
    for (int kk = 0; kk < Sk; kk += 32) {
        v16bf a, bfr;
#pragma unroll
        for (int e = 0; e < 16; ++e) a[e] = Ap[(size_t)row * Sk + kk + a_koff(e, g)];
#pragma unroll
        for (int e = 0; e < 16; ++e) bfr[e] = Bp[(size_t)(kk + 16 * g + e) * HDm + 16 * wave + col];
        c = __builtin_amdgcn_wmma_f32_16x16x32_bf16(false, a, false, bfr, (short)0, c, false, false);
    }
#pragma unroll
    for (int r = 0; r < 8; ++r)
        x1[((size_t)(b * NH + h) * Lq + 16 * mt + r + 8 * g) * HDm + 16 * wave + col] = c[r];
}

// ---------------- hacc = a @ Hv (16h x 512) @ (512 x 128), then combine with rv_w2/rv_b2 + x1 ----------------
__global__ void __launch_bounds__(256) k_hacc_combine(
    const float* __restrict__ xyz,
    const float* __restrict__ rv_w1, const float* __restrict__ rv_b1,
    const float* __restrict__ rv_w2, const float* __restrict__ rv_b2,
    const bf16* __restrict__ abf_mh, const float* __restrict__ x1,
    bf16* __restrict__ Xbf) {
    __shared__ bf16  sHv[NCH][HID];    // 16 KB
    __shared__ float sHacc[16][HID];   // 8 KB
    __shared__ float sW[HID][4];
    __shared__ float sXYZ[NCH][3];
    int tid = threadIdx.x, lane = tid & 31, wave = tid >> 5;
    int g = lane >> 4, col = lane & 15, row = lane & 15;
    int bm = blockIdx.x;
    int b = bm / Lq, m = bm % Lq;

    for (int j = tid; j < HID; j += 256) {
        sW[j][0] = rv_w1[j * 3 + 0];
        sW[j][1] = rv_w1[j * 3 + 1];
        sW[j][2] = rv_w1[j * 3 + 2];
        sW[j][3] = rv_b1[j];
    }
    __syncthreads();

    const bf16* Ap = abf_mh + (size_t)(b * Lq + m) * 16 * Sk;
    v8f c = {};
    for (int t = 0; t < Sk / NCH; ++t) {
        int n0 = t * NCH;
        for (int nn = tid; nn < NCH; nn += 256) {
            const float* p = xyz + ((size_t)(m * Sk + n0 + nn) * NB + b) * 3;
            sXYZ[nn][0] = p[0]; sXYZ[nn][1] = p[1]; sXYZ[nn][2] = p[2];
        }
        __syncthreads();
        for (int idx = tid; idx < NCH * HID; idx += 256) {
            int nn = idx / HID, j = idx % HID;
            const float* w = sW[j];
            float hsum = fmaf(sXYZ[nn][0], w[0],
                         fmaf(sXYZ[nn][1], w[1],
                         fmaf(sXYZ[nn][2], w[2], w[3])));
            sHv[nn][j] = f2bf(fmaxf(hsum, 0.f));
        }
        __syncthreads();
        // wave w owns j-tile w; accumulate over all chunks (K = n)
#pragma unroll
        for (int kk = 0; kk < NCH; kk += 32) {
            v16bf a, bfr;
#pragma unroll
            for (int e = 0; e < 16; ++e) a[e] = Ap[(size_t)row * Sk + n0 + kk + a_koff(e, g)];
#pragma unroll
            for (int e = 0; e < 16; ++e) bfr[e] = sHv[kk + 16 * g + e][16 * wave + col];
            c = __builtin_amdgcn_wmma_f32_16x16x32_bf16(false, a, false, bfr, (short)0, c, false,
                                                        false);
        }
        __syncthreads();
    }
#pragma unroll
    for (int r = 0; r < 8; ++r) sHacc[r + 8 * g][16 * wave + col] = c[r];
    __syncthreads();
    // combine: x = x1 + rv_w2_slice @ hacc + rv_b2_slice   (softmax row-sum == 1)
    {
        int h = tid >> 5;       // 0..7
        int cc = tid & 31;      // 0..31
        float acc = x1[((size_t)(b * NH + h) * Lq + m) * HDm + cc];
        const float* w2 = rv_w2 + (size_t)(h * HDm + cc) * HID;
        for (int j = 0; j < HID; ++j) acc = fmaf(w2[j], sHacc[h][j], acc);
        acc += rv_b2[h * HDm + cc];
        Xbf[((size_t)(m * NB) + b) * DIMk + h * HDm + cc] = f2bf(acc);
    }
}

// ---------------- host-side orchestration ----------------
extern "C" void kernel_launch(void* const* d_in, const int* in_sizes, int n_in,
                              void* d_out, int out_size, void* d_ws, size_t ws_size,
                              hipStream_t stream) {
    const float* q      = (const float*)d_in[0];
    const float* k      = (const float*)d_in[1];
    const float* v      = (const float*)d_in[2];
    const float* xyz    = (const float*)d_in[3];
    const float* in_w   = (const float*)d_in[4];
    const float* in_b   = (const float*)d_in[5];
    const float* out_w  = (const float*)d_in[6];
    const float* out_b  = (const float*)d_in[7];
    const float* rq_w1  = (const float*)d_in[8];
    const float* rq_b1  = (const float*)d_in[9];
    const float* rq_w2  = (const float*)d_in[10];
    const float* rq_b2  = (const float*)d_in[11];
    const float* rk_w1  = (const float*)d_in[12];
    const float* rk_b1  = (const float*)d_in[13];
    const float* rk_w2  = (const float*)d_in[14];
    const float* rk_b2  = (const float*)d_in[15];
    const float* rv_w1  = (const float*)d_in[16];
    const float* rv_b1  = (const float*)d_in[17];
    const float* rv_w2  = (const float*)d_in[18];
    const float* rv_b2  = (const float*)d_in[19];

    const float scale = 0.17677669529663687f;  // 32^-0.5

    char* ws = (char*)d_ws;
    auto alloc = [&](size_t bytes) -> void* {
        void* p = (void*)ws;
        ws += (bytes + 255) & ~(size_t)255;
        return p;
    };
    bf16*  qbf    = (bf16*)alloc((size_t)Lq * NB * DIMk * 2);
    bf16*  kbf    = (bf16*)alloc((size_t)Sk * NB * DIMk * 2);
    bf16*  vbf    = (bf16*)alloc((size_t)Sk * NB * DIMk * 2);
    bf16*  wbf    = (bf16*)alloc((size_t)3 * DIMk * DIMk * 2);
    bf16*  owbf   = (bf16*)alloc((size_t)DIMk * DIMk * 2);
    float* qflat  = (float*)alloc((size_t)Lq * NB * DIMk * 4);
    float* kflat  = (float*)alloc((size_t)Sk * NB * DIMk * 4);
    float* vflat  = (float*)alloc((size_t)Sk * NB * DIMk * 4);
    float* qs     = (float*)alloc((size_t)NB * NH * Lq * HDm * 4);
    float* khf    = (float*)alloc((size_t)NB * NH * Sk * HDm * 4);
    bf16*  vhb    = (bf16*)alloc((size_t)NB * NH * Sk * HDm * 2);
    bf16*  QW     = (bf16*)alloc((size_t)NB * Lq * 16 * HID * 2);
    float* qb     = (float*)alloc((size_t)NB * Lq * NH * 4);
    float* KW     = (float*)alloc((size_t)NB * Sk * NH * HID * 4);
    float* kb     = (float*)alloc((size_t)NB * NH * Sk * 4);
    bf16*  abf_hm = (bf16*)alloc((size_t)NB * NH * Lq * Sk * 2);
    bf16*  abf_mh = (bf16*)alloc((size_t)NB * Lq * 16 * Sk * 2);
    float* x1     = (float*)alloc((size_t)NB * NH * Lq * HDm * 4);
    bf16*  Xbf    = (bf16*)alloc((size_t)Lq * NB * DIMk * 2);

    auto cvt = [&](const float* s, bf16* d, int n) {
        k_f32_to_bf16<<<dim3((n + 255) / 256), dim3(256), 0, stream>>>(s, d, n);
    };
    cvt(q, qbf, Lq * NB * DIMk);
    cvt(k, kbf, Sk * NB * DIMk);
    cvt(v, vbf, Sk * NB * DIMk);
    cvt(in_w, wbf, 3 * DIMk * DIMk);
    cvt(out_w, owbf, DIMk * DIMk);

    // in-projections (WMMA GEMMs): flat(t*B+b, 256) @ W^T + bias
    k_gemm_bf16<<<dim3(DIMk / 16, (Lq * NB) / 16), dim3(32), 0, stream>>>(
        qbf, wbf, in_b, qflat, Lq * NB, DIMk, DIMk);
    k_gemm_bf16<<<dim3(DIMk / 16, (Sk * NB) / 16), dim3(32), 0, stream>>>(
        kbf, wbf + (size_t)DIMk * DIMk, in_b + DIMk, kflat, Sk * NB, DIMk, DIMk);
    k_gemm_bf16<<<dim3(DIMk / 16, (Sk * NB) / 16), dim3(32), 0, stream>>>(
        vbf, wbf + (size_t)2 * DIMk * DIMk, in_b + 2 * DIMk, vflat, Sk * NB, DIMk, DIMk);

    k_reshape_heads<<<dim3(512), dim3(256), 0, stream>>>(qflat, qs, nullptr, Lq, scale);
    k_reshape_heads<<<dim3(1024), dim3(256), 0, stream>>>(kflat, khf, nullptr, Sk, 1.0f);
    k_reshape_heads<<<dim3(1024), dim3(256), 0, stream>>>(vflat, nullptr, vhb, Sk, 1.0f);

    k_qw<<<dim3(2048), dim3(256), 0, stream>>>(qs, rq_w2, QW);
    k_qb<<<dim3(16), dim3(256), 0, stream>>>(qs, rq_b2, qb);
    k_kw<<<dim3(4096), dim3(256), 0, stream>>>(khf, rk_w2, KW, scale);
    k_kb<<<dim3(32), dim3(256), 0, stream>>>(khf, rk_b2, kb, scale);

    k_logits_softmax<<<dim3(NB * Lq), dim3(256), 0, stream>>>(
        xyz, rq_w1, rq_b1, rk_w1, rk_b1, QW, qb, KW, kb, qs, khf, abf_hm, abf_mh);

    k_x1<<<dim3(NB * NH * (Lq / 16)), dim3(64), 0, stream>>>(abf_hm, vhb, x1);

    k_hacc_combine<<<dim3(NB * Lq), dim3(256), 0, stream>>>(
        xyz, rv_w1, rv_b1, rv_w2, rv_b2, abf_mh, x1, Xbf);

    // out-projection (WMMA GEMM) directly into d_out: (l*B+b, 256)
    k_gemm_bf16<<<dim3(DIMk / 16, (Lq * NB) / 16), dim3(32), 0, stream>>>(
        Xbf, owbf, out_b, (float*)d_out, Lq * NB, DIMk, DIMk);

    (void)in_sizes; (void)n_in; (void)out_size; (void)ws_size;
}
